// CrossAttention_50440095924277
// MI455X (gfx1250) — matrix operand order repaired and checked by reading, not compiled, not run on hardware
//
#include <hip/hip_runtime.h>
#include <hip/hip_bf16.h>
#include <stdint.h>

// Problem constants (match reference)
#define B_   4
#define T_   512
#define TS_  512
#define J_   17
#define D_   256
#define H_   8
#define CD_  32
#define NS_  (B_ * J_ * H_)     // 544 attention slices
#define M_   (B_ * T_ * J_)     // 34816 GEMM rows
#define SCALE_ 0.17677669529663687f   // 1/sqrt(32)

typedef __attribute__((ext_vector_type(16))) __bf16 v16bf;
typedef __attribute__((ext_vector_type(2)))  __bf16 v2bf;
typedef __attribute__((ext_vector_type(8)))  float  v8f;
typedef unsigned short u16;
typedef unsigned int   u32;
typedef unsigned long long u64;

// 16 bf16 elements = one WMMA A/B fragment per lane (8 VGPRs)
union Frag16 {
  u16   u[16];
  uint4 q[2];
  v16bf v;
};

__device__ __forceinline__ u16 f2bf(float f) {
  // round-to-nearest-even f32 -> bf16
  u32 u = __float_as_uint(f);
  u32 r = u + 0x7FFFu + ((u >> 16) & 1u);
  return (u16)(r >> 16);
}

__device__ __forceinline__ u32 pack2bf(float lo, float hi) {
#if __has_builtin(__builtin_amdgcn_cvt_pk_bf16_f32)
  v2bf t = __builtin_amdgcn_cvt_pk_bf16_f32(lo, hi);
  return __builtin_bit_cast(u32, t);
#else
  return (u32)f2bf(lo) | ((u32)f2bf(hi) << 16);
#endif
}

__device__ __forceinline__ uint4 pack8bf(float4 a, float4 b) {
  uint4 r;
  r.x = pack2bf(a.x, a.y);
  r.y = pack2bf(a.z, a.w);
  r.z = pack2bf(b.x, b.y);
  r.w = pack2bf(b.z, b.w);
  return r;
}

__device__ __forceinline__ v8f wmma_bf16(const Frag16& a, const Frag16& b, v8f c) {
  return __builtin_amdgcn_wmma_f32_16x16x32_bf16(false, a.v, false, b.v,
                                                 (short)0, c, false, false);
}

// Async 16B global -> LDS copy (VGLOBAL GLOBAL_LOAD_ASYNC_TO_LDS_B128, ASYNCcnt)
__device__ __forceinline__ void async_g2l_b128(const void* gaddr, u32 ldsoff) {
  asm volatile("global_load_async_to_lds_b128 %0, %1, off"
               :: "v"(ldsoff), "v"((u64)(uintptr_t)gaddr) : "memory");
}
__device__ __forceinline__ void wait_async0() {
  asm volatile("s_wait_asynccnt 0x0" ::: "memory");
}

// ---------------------------------------------------------------------------
// Kernel 0: one-shot f32 -> bf16 conversion (weights). 8 elements/thread.
// ---------------------------------------------------------------------------
__global__ __launch_bounds__(256) void k_cvt(const float* __restrict__ src,
                                             u16* __restrict__ dst, int n8) {
  int i = blockIdx.x * 256 + threadIdx.x;
  if (i >= n8) return;
  const float4* s = reinterpret_cast<const float4*>(src) + (size_t)i * 2;
  float4 a = s[0], b = s[1];
  reinterpret_cast<uint4*>(dst)[i] = pack8bf(a, b);
}

// ---------------------------------------------------------------------------
// Kernel 1: Q/K/V projections.  out[m][n] = sum_k x[m][k] * W[n][k]
// Block = 8 waves, 64 t-rows x 256 n-cols. Phase 1 converts the 64x256
// activation rows ONCE into an LDS bf16 tile (stride 264 halves: conflict-free
// A-fragment reads). Phase 2: wave (wgrp,nhalf) runs 8 n-tiles x 8 k-step
// WMMAs, A from LDS, B from pre-converted bf16 weights.
//   proj 0: Q -> Qp[slice][t][32]
//   proj 1: K -> Kp[slice][s][32]
//   proj 2: V -> Vt[slice][32][TS]  (transposed so PV B-fragments are contiguous)
// ---------------------------------------------------------------------------
#define ALDS_STRIDE 264   // 256 + 8 halves pad -> 132 dwords, rows hit distinct banks

__global__ __launch_bounds__(256) void k_proj(
    const float* __restrict__ q, const float* __restrict__ kv,
    const u16* __restrict__ Wb,        // 4 x [256][256] bf16 (q,k,v,p)
    u16* __restrict__ Qp, u16* __restrict__ Kp, u16* __restrict__ Vt)
{
  __shared__ __align__(16) u16 Alds[64 * ALDS_STRIDE];   // ~33 KB

  const int proj = blockIdx.z;
  const float* x = (proj == 0) ? q : kv;
  const u16* W = Wb + (size_t)proj * D_ * D_;

  const int tchunk = blockIdx.x & 7;          // 8 chunks of 64 t-rows
  const int bj     = blockIdx.x >> 3;         // 0..67
  const int b = bj / J_, j = bj % J_;
  const int tid = threadIdx.x;
  const int wave = tid >> 5, lane = tid & 31;
  const int lhalf = (lane < 16) ? 0 : 1;
  const int l16   = lane & 15;
  const int wgrp  = wave & 3;                 // 16-row group
  const int nhalf = wave >> 2;                // 128-col half
  const int tblk  = tchunk * 64;

  // ---- Phase 1: stage + convert A tile: thread -> (row, 64-col segment)
  {
    const int row = tid >> 2;
    const int seg = (tid & 3) * 64;
    const float* srow =
        x + ((size_t)(b * T_ + tblk + row) * J_ + j) * D_ + seg;
    u16* drow = Alds + row * ALDS_STRIDE + seg;
#pragma unroll
    for (int i = 0; i < 8; ++i) {
      const float4* s4 = reinterpret_cast<const float4*>(srow + i * 8);
      float4 a = s4[0], bb = s4[1];
      *reinterpret_cast<uint4*>(drow + i * 8) = pack8bf(a, bb);
    }
  }
  __syncthreads();

  const u16* arow = Alds + (wgrp * 16 + l16) * ALDS_STRIDE;
  const int t0 = tblk + wgrp * 16;

  // ---- Phase 2: 8 n-tiles x 8 k-steps of WMMA
#pragma unroll
  for (int nt = 0; nt < 8; ++nt) {
    const int n0  = nhalf * 128 + nt * 16;
    const int h   = n0 >> 5;
    const int ncd = n0 & 31;
    const u16* brow = W + (size_t)(n0 + l16) * D_;

    v8f acc = {};
#pragma unroll
    for (int kk = 0; kk < D_; kk += 32) {
      Frag16 a, bf;
      // A layout: lanes 0-15 K kk..kk+15; lanes 16-31 K kk+16..kk+31
      const uint4* pa = reinterpret_cast<const uint4*>(arow + kk + (lhalf ? 16 : 0));
      a.q[0] = pa[0];
      a.q[1] = pa[1];
      // B layout: lanes 0-15 K kk..kk+15 of row n0+l16; lanes 16-31 K kk+16..kk+31
      const uint4* pb = reinterpret_cast<const uint4*>(brow + kk + (lhalf ? 16 : 0));
      bf.q[0] = pb[0];
      bf.q[1] = pb[1];
      acc = wmma_bf16(a, bf, acc);
    }

    const int slice = (b * J_ + j) * H_ + h;
    // C layout: VGPR r, lanes 0-15 -> (M=r, N=l16); lanes 16-31 -> (M=r+8, N=l16)
    if (proj < 2) {
      u16* dst = (proj == 0 ? Qp : Kp) + (size_t)slice * T_ * CD_;
#pragma unroll
      for (int r = 0; r < 8; ++r) {
        int t  = t0 + r + lhalf * 8;
        dst[(size_t)t * CD_ + ncd + l16] = f2bf(acc[r]);
      }
    } else {
      u16* dst = Vt + (size_t)slice * CD_ * TS_;
#pragma unroll
      for (int r = 0; r < 8; ++r) {
        int s = t0 + r + lhalf * 8;
        dst[(size_t)(ncd + l16) * TS_ + s] = f2bf(acc[r]);
      }
    }
  }
}

// ---------------------------------------------------------------------------
// Kernel 2: flash attention per slice. Block = 8 waves x 16 q-rows = 128 rows.
// K/V are staged chunk-by-chunk (128 keys) into double-buffered LDS with
// GLOBAL_LOAD_ASYNC_TO_LDS_B128 (ASYNCcnt), overlapping the copy of chunk c+1
// with WMMA+softmax on chunk c. LDS strides padded for conflict-free reads:
//   K: 40 halves/row (r*20 mod 64 banks distinct), V: 136 halves/row (r*4).
// P tile bounces through a wave-private LDS buffer (C-layout -> A-layout).
// ---------------------------------------------------------------------------
#define KSTR 40     // K LDS row stride in halves (128 rows x 40 = 10240 B)
#define VSTR 136    // V LDS row stride in halves (32 rows x 136 = 8704 B)
#define SCHUNK 128  // keys per staged chunk

__global__ __launch_bounds__(256) void k_attn(
    const u16* __restrict__ Qp, const u16* __restrict__ Kp,
    const u16* __restrict__ Vt, u16* __restrict__ AO)
{
  __shared__ __align__(16) u16 Kl[2][SCHUNK * KSTR];
  __shared__ __align__(16) u16 Vl[2][CD_ * VSTR];
  __shared__ __align__(16) u16 Pt[8 * 16 * 32];

  const int slice = blockIdx.y;
  const int h = slice % H_;
  const int j = (slice / H_) % J_;
  const int b = slice / (H_ * J_);
  const int tid  = threadIdx.x;
  const int wave = tid >> 5, lane = tid & 31;
  const int lhalf = (lane < 16) ? 0 : 1;
  const int l16   = lane & 15;
  const int t0 = blockIdx.x * 128 + wave * 16;

  const u16* Qs = Qp + (size_t)slice * T_ * CD_;
  const u16* Ks = Kp + (size_t)slice * TS_ * CD_;
  const u16* Vs = Vt + (size_t)slice * CD_ * TS_;
  u16* Pw = Pt + wave * (16 * 32);

  // Issue async copies for one 128-key chunk: 512 x 16B (K) + 512 x 16B (V),
  // 4 transfers per thread.
  auto issue_chunk = [&](int s0, int buf) {
#pragma unroll
    for (int it = 0; it < 2; ++it) {
      int i = tid + it * 256;                    // 0..511
      {  // K chunk: row = i/4 (0..127), 16B piece = i%4
        int row = i >> 2, col = (i & 3) * 8;     // col in halves
        async_g2l_b128(Ks + (size_t)(s0 + row) * CD_ + col,
                       (u32)(uintptr_t)&Kl[buf][row * KSTR + col]);
      }
      {  // V chunk: row cd = i/16 (0..31), 16B piece = i%16
        int row = i >> 4, col = (i & 15) * 8;
        async_g2l_b128(Vs + (size_t)row * TS_ + s0 + col,
                       (u32)(uintptr_t)&Vl[buf][row * VSTR + col]);
      }
    }
  };

  issue_chunk(0, 0);

  // Q fragment (A layout), overlapped with the first async copy
  Frag16 qa;
  {
    const uint4* p = reinterpret_cast<const uint4*>(
        Qs + (size_t)(t0 + l16) * CD_ + (lhalf ? 16 : 0));
    qa.q[0] = p[0];
    qa.q[1] = p[1];   // +8 halves
  }

  float m[8], l[8];
  v8f O0 = {}, O1 = {};
#pragma unroll
  for (int r = 0; r < 8; ++r) { m[r] = -__builtin_inff(); l[r] = 0.0f; }

  wait_async0();
  __syncthreads();

  for (int c = 0; c < TS_ / SCHUNK; ++c) {
    const int buf = c & 1;
    if (c + 1 < TS_ / SCHUNK) issue_chunk((c + 1) * SCHUNK, buf ^ 1);
    const u16* Kb = &Kl[buf][0];
    const u16* Vb = &Vl[buf][0];

#pragma unroll
    for (int sl = 0; sl < SCHUNK; sl += 32) {
      // K^T B-fragments from LDS: column N = key index, K dim = head dim
      Frag16 kb0, kb1;
      {
        const uint4* p0 = reinterpret_cast<const uint4*>(
            Kb + (sl + l16) * KSTR + (lhalf ? 16 : 0));
        kb0.q[0] = p0[0]; kb0.q[1] = p0[1];
        const uint4* p1 = reinterpret_cast<const uint4*>(
            Kb + (sl + 16 + l16) * KSTR + (lhalf ? 16 : 0));
        kb1.q[0] = p1[0]; kb1.q[1] = p1[1];
      }
      v8f z = {};
      v8f S0 = wmma_bf16(qa, kb0, z);
      v8f S1 = wmma_bf16(qa, kb1, z);

      // Online softmax. Row r lives in (VGPR r, 16-lane half); xor-butterfly
      // masks 1,2,4,8 reduce within each half.
#pragma unroll
      for (int r = 0; r < 8; ++r) {
        float s0 = S0[r] * SCALE_;
        float s1 = S1[r] * SCALE_;
        float rm = fmaxf(s0, s1);
        rm = fmaxf(rm, __shfl_xor(rm, 1, 32));
        rm = fmaxf(rm, __shfl_xor(rm, 2, 32));
        rm = fmaxf(rm, __shfl_xor(rm, 4, 32));
        rm = fmaxf(rm, __shfl_xor(rm, 8, 32));
        float mnew = fmaxf(m[r], rm);
        float corr = __expf(m[r] - mnew);
        float p0 = __expf(s0 - mnew);
        float p1 = __expf(s1 - mnew);
        float rs = p0 + p1;
        rs += __shfl_xor(rs, 1, 32);
        rs += __shfl_xor(rs, 2, 32);
        rs += __shfl_xor(rs, 4, 32);
        rs += __shfl_xor(rs, 8, 32);
        l[r] = l[r] * corr + rs;
        m[r] = mnew;
        O0[r] *= corr;
        O1[r] *= corr;
        int row = r + lhalf * 8;
        Pw[row * 32 + l16]      = f2bf(p0);
        Pw[row * 32 + 16 + l16] = f2bf(p1);
      }

      // Wave-private LDS round-trip: DS ops are in-order per wave.
      asm volatile("s_wait_dscnt 0x0" ::: "memory");

      // P as A fragment (16 q-rows x 32 keys)
      Frag16 pa;
      {
        const uint4* p = reinterpret_cast<const uint4*>(Pw + l16 * 32 + (lhalf ? 16 : 0));
        pa.q[0] = p[0];
        pa.q[1] = p[1];
      }
      // V B-fragments from LDS: Vl is [cd][s], 16 contiguous halves per lane
      Frag16 vb0, vb1;
      {
        const uint4* p0 = reinterpret_cast<const uint4*>(
            Vb + l16 * VSTR + sl + (lhalf ? 16 : 0));
        vb0.q[0] = p0[0]; vb0.q[1] = p0[1];
        const uint4* p1 = reinterpret_cast<const uint4*>(
            Vb + (16 + l16) * VSTR + sl + (lhalf ? 16 : 0));
        vb1.q[0] = p1[0]; vb1.q[1] = p1[1];
      }
      O0 = wmma_bf16(pa, vb0, O0);
      O1 = wmma_bf16(pa, vb1, O1);
    }

    // All waves must finish this chunk's async copies (their own) and the
    // whole block must pass the barrier before the buffer is reused.
    wait_async0();
    __syncthreads();
  }

  // Normalize and store attention output as bf16 in [b][t][j][d] layout
#pragma unroll
  for (int r = 0; r < 8; ++r) {
    float inv = 1.0f / l[r];
    int t = t0 + r + lhalf * 8;
    size_t rowoff = ((size_t)(b * T_ + t) * J_ + j) * D_ + h * CD_;
    AO[rowoff + l16]      = f2bf(O0[r] * inv);
    AO[rowoff + 16 + l16] = f2bf(O1[r] * inv);
  }
}

// ---------------------------------------------------------------------------
// Kernel 3: output projection. out[m][n] = sum_k AO[m][k]*Wpb[n][k] + bp[n]
// Both operands already bf16 -> pure b128 loads + WMMA.
// ---------------------------------------------------------------------------
__global__ __launch_bounds__(256) void k_out(
    const u16* __restrict__ AO, const u16* __restrict__ Wpb,
    const float* __restrict__ bp, float* __restrict__ out)
{
  const int wave = threadIdx.x >> 5, lane = threadIdx.x & 31;
  const int lhalf = (lane < 16) ? 0 : 1;
  const int l16   = lane & 15;
  const int m0 = blockIdx.x * 128 + wave * 16;
  const int n0 = blockIdx.y * 16;

  const u16* arow = AO  + (size_t)(m0 + l16) * D_;
  const u16* brow = Wpb + (size_t)(n0 + l16) * D_;

  v8f acc = {};
#pragma unroll
  for (int kk = 0; kk < D_; kk += 32) {
    Frag16 a, bf;
    const uint4* pa = reinterpret_cast<const uint4*>(arow + kk + (lhalf ? 16 : 0));
    a.q[0] = pa[0];
    a.q[1] = pa[1];
    const uint4* pb = reinterpret_cast<const uint4*>(brow + kk + (lhalf ? 16 : 0));
    bf.q[0] = pb[0];
    bf.q[1] = pb[1];
    acc = wmma_bf16(a, bf, acc);
  }

  float bias = bp[n0 + l16];
#pragma unroll
  for (int r = 0; r < 8; ++r) {
    int mr = m0 + r + lhalf * 8;
    out[(size_t)mr * D_ + n0 + l16] = acc[r] + bias;
  }
}

// ---------------------------------------------------------------------------
extern "C" void kernel_launch(void* const* d_in, const int* in_sizes, int n_in,
                              void* d_out, int out_size, void* d_ws, size_t ws_size,
                              hipStream_t stream) {
  const float* q  = (const float*)d_in[0];
  const float* kv = (const float*)d_in[1];
  const float* Wq = (const float*)d_in[2];
  const float* Wk = (const float*)d_in[3];
  const float* Wv = (const float*)d_in[4];
  const float* Wp = (const float*)d_in[5];
  const float* bp = (const float*)d_in[6];
  float* out = (float*)d_out;

  // Workspace layout (bf16): Wb(4x64K) | Qp | Kp | Vt | AO  (~68.5 MB total)
  char* ws = (char*)d_ws;
  const size_t wseg = (size_t)D_ * D_ * sizeof(u16);          // 131,072 B
  const size_t seg  = (size_t)NS_ * T_ * CD_ * sizeof(u16);   // 17,825,792 B
  u16* Wb = (u16*)(ws);                       // [Wq|Wk|Wv|Wp] bf16
  u16* Qp = (u16*)(ws + 4 * wseg);
  u16* Kp = (u16*)(ws + 4 * wseg + seg);
  u16* Vt = (u16*)(ws + 4 * wseg + 2 * seg);
  u16* AO = (u16*)(ws + 4 * wseg + 3 * seg);

  // 0) weights -> bf16 (65536 elems each, 8 per thread)
  const int n8 = (D_ * D_) / 8;               // 8192 threads -> 32 blocks
  k_cvt<<<dim3(32), 256, 0, stream>>>(Wq, Wb + 0 * D_ * D_, n8);
  k_cvt<<<dim3(32), 256, 0, stream>>>(Wk, Wb + 1 * D_ * D_, n8);
  k_cvt<<<dim3(32), 256, 0, stream>>>(Wv, Wb + 2 * D_ * D_, n8);
  k_cvt<<<dim3(32), 256, 0, stream>>>(Wp, Wb + 3 * D_ * D_, n8);

  // 1) projections: grid.x = (T/64)*(B*J) = 544, grid.z = {Q,K,V}
  k_proj<<<dim3(544, 1, 3), 256, 0, stream>>>(q, kv, Wb, Qp, Kp, Vt);
  // 2) flash attention: 4 q-chunks x 544 slices
  k_attn<<<dim3(4, NS_), 256, 0, stream>>>(Qp, Kp, Vt, AO);
  // 3) output projection: 34816/128 = 272 row-chunks x 16 col-tiles
  k_out<<<dim3(272, 16), 256, 0, stream>>>(AO, Wb + 3 * D_ * D_, bp, out);
}
